// SpectralFRR_32315333935572
// MI455X (gfx1250) — compile-verified, matrix operand order checked by reading, not run on hardware
//
#include <hip/hip_runtime.h>

// ---------------------------------------------------------------------------
// CDNA5 / gfx1250 implementation of the SpectralFRR reference.
// Residual stream f32; all GEMMs via v_wmma_f32_16x16x32_bf16 with
// double-buffered LDS and async global->LDS staging (ASYNCcnt-tracked).
// Attention = batched WMMA GEMMs + causal softmax.
// ---------------------------------------------------------------------------

typedef __attribute__((ext_vector_type(16))) __bf16 v16bf;
typedef __attribute__((ext_vector_type(8)))  __bf16 v8bf;
typedef __attribute__((ext_vector_type(8)))  float  v8f;
typedef int v4i_ __attribute__((vector_size(16)));   // matches builtin's V4i

namespace cfg {
constexpr int B_ = 2, T_ = 512, D_ = 1024, H_ = 16, HD_ = 64, FF_ = 4096, V_ = 32000;
constexpr int NTOK = B_ * T_;                 // 1024
constexpr int NSC = 4, ITERS = 7, NL = NSC * ITERS, NM = 4, RK = 8;
constexpr float EPS_ = 1.1920929e-07f;
}
using namespace cfg;

// ---------------- CDNA5 async global->LDS staging ----------------

#if defined(__has_builtin)
#if __has_builtin(__builtin_amdgcn_global_load_async_to_lds_b128)
#define USE_ASYNC_LDS 1
#endif
#if __has_builtin(__builtin_amdgcn_s_wait_asynccnt)
#define HAVE_WAIT_ASYNC_BUILTIN 1
#endif
#endif

// copy 16 bytes global -> LDS (async when supported)
__device__ __forceinline__ void stage16(const __bf16* g, __bf16* l) {
#ifdef USE_ASYNC_LDS
  // prototype (from toolchain diagnostic): (v4i __device__*, v4i __shared__*, imm, imm)
  // generic pointers convert implicitly to CUDA-AS-qualified builtin params.
  __builtin_amdgcn_global_load_async_to_lds_b128((v4i_*)g, (v4i_*)l, 0, 0);
#else
  *(uint4*)l = *(const uint4*)g;
#endif
}

// wait for this wave's outstanding async LDS transfers
__device__ __forceinline__ void async_join() {
#ifdef USE_ASYNC_LDS
#ifdef HAVE_WAIT_ASYNC_BUILTIN
  __builtin_amdgcn_s_wait_asynccnt(0);
#else
  asm volatile("s_wait_asynccnt 0x0" ::: "memory");
#endif
#endif
}

// ---------------- device helpers ----------------

__device__ __forceinline__ __bf16 f2bf(float f) {
  unsigned u = __builtin_bit_cast(unsigned, f);
  u += 0x7FFFu + ((u >> 16) & 1u);            // round-to-nearest-even
  unsigned short h = (unsigned short)(u >> 16);
  return __builtin_bit_cast(__bf16, h);
}

__device__ __forceinline__ float wave_sum(float v) {
#pragma unroll
  for (int m = 16; m > 0; m >>= 1) v += __shfl_xor(v, m, 32);
  return v;
}
__device__ __forceinline__ float wave_max(float v) {
#pragma unroll
  for (int m = 16; m > 0; m >>= 1) v = fmaxf(v, __shfl_xor(v, m, 32));
  return v;
}

// ---------------- elementwise / staging kernels ----------------

__global__ void cvt_bf16_kernel(const float* __restrict__ src,
                                __bf16* __restrict__ dst, int n) {
  int i = blockIdx.x * 256 + threadIdx.x;
  if (i < n) dst[i] = f2bf(src[i]);
}

__global__ void embed_kernel(const int* __restrict__ tok,
                             const float* __restrict__ ew,
                             float* __restrict__ x) {
  int i = blockIdx.x * 256 + threadIdx.x;      // over NTOK*D
  int t = i / D_, d = i - t * D_;
  x[i] = ew[(long)tok[t] * D_ + d];
}

// rms(x)*norm_w*gamma + beta -> bf16 ; one wave per token row
__global__ void rmsnorm_kernel(const float* __restrict__ x,
                               const float* __restrict__ nw,
                               const float* __restrict__ gamma,
                               const float* __restrict__ beta,
                               __bf16* __restrict__ out) {
  int row = blockIdx.x * 8 + (threadIdx.x >> 5);
  int lane = threadIdx.x & 31;
  const float* xr = x + (long)row * D_;
  float s = 0.f;
#pragma unroll
  for (int i = 0; i < D_ / 32; i++) { float v = xr[lane + i * 32]; s += v * v; }
  s = wave_sum(s);
  float r = rsqrtf(s * (1.0f / D_) + EPS_);
  __bf16* o = out + (long)row * D_;
#pragma unroll
  for (int i = 0; i < D_ / 32; i++) {
    int d = lane + i * 32;
    float g = gamma ? gamma[d] : 1.f;
    float bb = beta ? beta[d] : 0.f;
    o[d] = f2bf(xr[d] * r * nw[d] * g + bb);
  }
}

// softmax(route) blend of the 4 LoRA mode tensors (each H*HD*RK = 8192 floats)
__global__ void blendmodes_kernel(const float* __restrict__ route,
                                  const float* __restrict__ qd, const float* __restrict__ qu,
                                  const float* __restrict__ kd, const float* __restrict__ ku,
                                  float* __restrict__ bqd, float* __restrict__ bqu,
                                  float* __restrict__ bkd, float* __restrict__ bku) {
  int i = blockIdx.x * 256 + threadIdx.x;      // 4 * 8192
  int tsel = i >> 13;
  int e = i & 8191;
  float w[NM];
  float mx = route[0];
#pragma unroll
  for (int m = 1; m < NM; m++) mx = fmaxf(mx, route[m]);
  float sum = 0.f;
#pragma unroll
  for (int m = 0; m < NM; m++) { w[m] = __expf(route[m] - mx); sum += w[m]; }
  float inv = 1.f / sum;
  const float* src = (tsel == 0) ? qd : (tsel == 1) ? qu : (tsel == 2) ? kd : ku;
  float* dst = (tsel == 0) ? bqd : (tsel == 1) ? bqu : (tsel == 2) ? bkd : bku;
  float acc = 0.f;
#pragma unroll
  for (int m = 0; m < NM; m++) acc += w[m] * src[m * 8192 + e];
  dst[e] = acc * inv;
}

// qkv f32 (NTOK, 3*D) -> q,k with rank-8 LoRA delta -> bf16 (B,H,T,HD);
// v -> bf16 transposed (B,H,HD,T). One wave per (b,h,t).
__global__ void lora_stage_kernel(const float* __restrict__ qkv,
                                  const float* __restrict__ bqd, const float* __restrict__ bqu,
                                  const float* __restrict__ bkd, const float* __restrict__ bku,
                                  __bf16* __restrict__ qs, __bf16* __restrict__ ks,
                                  __bf16* __restrict__ vt) {
  int gid = blockIdx.x * 8 + (threadIdx.x >> 5);
  int lane = threadIdx.x & 31;
  int t = gid % T_;
  int h = (gid / T_) % H_;
  int b = gid / (T_ * H_);
  long rowb = (long)(b * T_ + t) * (3 * D_);
  int d0 = lane, d1 = lane + 32;
  long ob = (long)((b * H_ + h) * T_ + t) * HD_;

  // ---- q ----
  {
    float q0 = qkv[rowb + h * HD_ + d0], q1 = qkv[rowb + h * HD_ + d1];
    const float* Dm = bqd + h * HD_ * RK;   // [d][r]
    const float* Um = bqu + h * RK * HD_;   // [r][d]
    float rq[RK];
#pragma unroll
    for (int r = 0; r < RK; r++)
      rq[r] = wave_sum(q0 * Dm[d0 * RK + r] + q1 * Dm[d1 * RK + r]);
    float a0 = 0.f, a1 = 0.f;
#pragma unroll
    for (int r = 0; r < RK; r++) { a0 += rq[r] * Um[r * HD_ + d0]; a1 += rq[r] * Um[r * HD_ + d1]; }
    qs[ob + d0] = f2bf(q0 + a0);
    qs[ob + d1] = f2bf(q1 + a1);
  }
  // ---- k ----
  {
    float k0 = qkv[rowb + D_ + h * HD_ + d0], k1 = qkv[rowb + D_ + h * HD_ + d1];
    const float* Dm = bkd + h * HD_ * RK;
    const float* Um = bku + h * RK * HD_;
    float rk[RK];
#pragma unroll
    for (int r = 0; r < RK; r++)
      rk[r] = wave_sum(k0 * Dm[d0 * RK + r] + k1 * Dm[d1 * RK + r]);
    float a0 = 0.f, a1 = 0.f;
#pragma unroll
    for (int r = 0; r < RK; r++) { a0 += rk[r] * Um[r * HD_ + d0]; a1 += rk[r] * Um[r * HD_ + d1]; }
    ks[ob + d0] = f2bf(k0 + a0);
    ks[ob + d1] = f2bf(k1 + a1);
  }
  // ---- v (transposed stage for P@V as GEMM with (N,K) row-major B) ----
  {
    float v0 = qkv[rowb + 2 * D_ + h * HD_ + d0];
    float v1 = qkv[rowb + 2 * D_ + h * HD_ + d1];
    long vb = (long)(b * H_ + h) * HD_ * T_;
    vt[vb + (long)d0 * T_ + t] = f2bf(v0);
    vt[vb + (long)d1 * T_ + t] = f2bf(v1);
  }
}

// causal softmax over scores rows (len T); one wave per (b,h,t) row
__global__ void softmax_kernel(const float* __restrict__ S, __bf16* __restrict__ P) {
  int row = blockIdx.x * 8 + (threadIdx.x >> 5);
  int lane = threadIdx.x & 31;
  int t = row % T_;
  const float* sr = S + (long)row * T_;
  float vals[T_ / 32];
  float mx = -3.0e38f;
#pragma unroll
  for (int i = 0; i < T_ / 32; i++) {
    int s = lane + i * 32;
    float v = (s <= t) ? sr[s] * 0.125f : -3.0e38f;   // 1/sqrt(64)
    vals[i] = v;
    mx = fmaxf(mx, v);
  }
  mx = wave_max(mx);
  float sum = 0.f;
#pragma unroll
  for (int i = 0; i < T_ / 32; i++) { vals[i] = __expf(vals[i] - mx); sum += vals[i]; }
  sum = wave_sum(sum);
  float inv = 1.f / sum;
  __bf16* pr = P + (long)row * T_;
#pragma unroll
  for (int i = 0; i < T_ / 32; i++) pr[lane + i * 32] = f2bf(vals[i] * inv);
}

// (B,H,T,HD) f32 -> (B*T, D) bf16
__global__ void merge_heads_kernel(const float* __restrict__ ao, __bf16* __restrict__ o) {
  int i = blockIdx.x * 256 + threadIdx.x;      // NTOK*D, output index
  int b = i / (T_ * D_);
  int r = i - b * (T_ * D_);
  int t = r / D_;
  int d = r - t * D_;
  int h = d / HD_, hd = d - h * HD_;
  o[i] = f2bf(ao[(long)((b * H_ + h) * T_ + t) * HD_ + hd]);
}

__global__ void add_kernel(float* __restrict__ x, const float* __restrict__ y, int n) {
  int i = blockIdx.x * 256 + threadIdx.x;
  if (i < n) x[i] += y[i];
}

__global__ void silu_mul_kernel(const float* __restrict__ g, const float* __restrict__ u,
                                __bf16* __restrict__ o, int n) {
  int i = blockIdx.x * 256 + threadIdx.x;
  if (i < n) {
    float gv = g[i];
    float s = gv / (1.f + __expf(-gv));
    o[i] = f2bf(s * u[i]);
  }
}

// x = x_in + (x + dn - x_in) * iter_scale[li]
__global__ void iterblend_kernel(const float* __restrict__ x_in, const float* __restrict__ dn,
                                 float* __restrict__ x, const float* __restrict__ iter_s, int li) {
  float it = iter_s[li];
  int i = blockIdx.x * 256 + threadIdx.x;
  float xi = x_in[i];
  x[i] = xi + (x[i] + dn[i] - xi) * it;
}

// ---------------- WMMA GEMM: C(M,N) = A(M,K) @ B(N,K)^T, batched ----------------
// bf16 in, f32 out. Block tile BM x BN_, K-tile 64 (2 wmma K-steps),
// double-buffered LDS with async global->LDS staging, 8 waves,
// wave grid 4(M) x 2(N); per-wave 32 x BN_/2.

constexpr int BM = 128, BK = 64;

template <int BN_>
__global__ __launch_bounds__(256)
void gemm_bf16_kernel(const __bf16* __restrict__ A, int lda, long strideA,
                      const __bf16* __restrict__ Bm, int ldb, long strideB,
                      float* __restrict__ C, int ldc, long strideC,
                      int M, int N, int K) {
  constexpr int NTN = BN_ / 32;                 // n-tiles per wave (4 or 2)
  constexpr int ACH = BM * BK / 8 / 256;        // A 16B-chunks per thread (4)
  constexpr int BCH = BN_ * BK / 8 / 256;       // B chunks per thread (4 or 2)
  __shared__ __bf16 As[2][BM * BK];
  __shared__ __bf16 Bs[2][BN_ * BK];

  const int tid = threadIdx.x;
  const int lane = tid & 31, wave = tid >> 5;
  const int waveM = wave & 3, waveN = wave >> 2;   // 4 x 2 wave grid
  const int half = lane >> 4, lr = lane & 15;
  const int m0 = blockIdx.y * BM, n0 = blockIdx.x * BN_;

  const __bf16* Ag = A + (long)blockIdx.z * strideA;
  const __bf16* Bg = Bm + (long)blockIdx.z * strideB;
  float* Cg = C + (long)blockIdx.z * strideC;

  auto stage_tile = [&](int buf, int k0) {
#pragma unroll
    for (int c = 0; c < ACH; c++) {
      int chunk = tid + c * 256;                // 0..1023
      int row = chunk >> 3;                     // 0..127
      int col = (chunk & 7) * 8;                // 0..56
      int gr = m0 + row; if (gr >= M) gr = M - 1;
      stage16(&Ag[(long)gr * lda + k0 + col], &As[buf][row * BK + col]);
    }
#pragma unroll
    for (int c = 0; c < BCH; c++) {
      int chunk = tid + c * 256;
      int row = chunk >> 3;
      int col = (chunk & 7) * 8;
      int gn = n0 + row; if (gn >= N) gn = N - 1;
      stage16(&Bg[(long)gn * ldb + k0 + col], &Bs[buf][row * BK + col]);
    }
  };

  v8f acc[2][NTN];
#pragma unroll
  for (int i = 0; i < 2; i++)
#pragma unroll
    for (int j = 0; j < NTN; j++) acc[i][j] = (v8f){0.f, 0.f, 0.f, 0.f, 0.f, 0.f, 0.f, 0.f};

  const int nk = K / BK;
  stage_tile(0, 0);

  for (int kt = 0; kt < nk; kt++) {
    const int buf = kt & 1;
    async_join();          // this wave's staging of `buf` complete
    __syncthreads();       // everyone's staging of `buf` visible; prev reads done
    if (kt + 1 < nk) stage_tile(buf ^ 1, (kt + 1) * BK);   // overlap with compute

#pragma unroll
    for (int ks = 0; ks < 2; ks++) {
      const int cb = ks * 32 + half * 8;
      v16bf af[2], bfm[NTN];
#pragma unroll
      for (int mi = 0; mi < 2; mi++) {
        const __bf16* ptr = &As[buf][(waveM * 32 + mi * 16 + lr) * BK + cb];
        v8bf lo = *(const v8bf*)ptr;
        v8bf hi = *(const v8bf*)(ptr + 16);
        af[mi] = __builtin_shufflevector(lo, hi, 0, 1, 2, 3, 4, 5, 6, 7,
                                         8, 9, 10, 11, 12, 13, 14, 15);
      }
#pragma unroll
      for (int ni = 0; ni < NTN; ni++) {
        const __bf16* ptr = &Bs[buf][(waveN * (BN_ / 2) + ni * 16 + lr) * BK + cb];
        v8bf lo = *(const v8bf*)ptr;
        v8bf hi = *(const v8bf*)(ptr + 16);
        bfm[ni] = __builtin_shufflevector(lo, hi, 0, 1, 2, 3, 4, 5, 6, 7,
                                          8, 9, 10, 11, 12, 13, 14, 15);
      }
#pragma unroll
      for (int mi = 0; mi < 2; mi++)
#pragma unroll
        for (int ni = 0; ni < NTN; ni++)
          acc[mi][ni] = __builtin_amdgcn_wmma_f32_16x16x32_bf16(
              false, af[mi], false, bfm[ni], (short)0, acc[mi][ni], false, false);
    }
  }

  // store: C/D layout: vgpr r -> row (r + half*8), col lr within 16x16 tile
#pragma unroll
  for (int mi = 0; mi < 2; mi++) {
    int rbase = m0 + waveM * 32 + mi * 16 + half * 8;
#pragma unroll
    for (int ni = 0; ni < NTN; ni++) {
      int col = n0 + waveN * (BN_ / 2) + ni * 16 + lr;
      if (col < N) {
#pragma unroll
        for (int r = 0; r < 8; r++) {
          int row = rbase + r;
          if (row < M) Cg[(long)row * ldc + col] = acc[mi][ni][r];
        }
      }
    }
  }
}

// ---------------- host-side orchestration ----------------

static inline int ceil_div(int a, int b) { return (a + b - 1) / b; }

extern "C" void kernel_launch(void* const* d_in, const int* in_sizes, int n_in,
                              void* d_out, int out_size, void* d_ws, size_t ws_size,
                              hipStream_t stream) {
  (void)in_sizes; (void)n_in; (void)out_size; (void)ws_size;
  const int*   tokens  = (const int*)d_in[0];
  const float* embed_w = (const float*)d_in[1];
  const float* lm_w    = (const float*)d_in[2];
  const float* norm_w  = (const float*)d_in[3];
  const float* norm1_w = (const float*)d_in[4];
  const float* norm2_w = (const float*)d_in[5];
  const float* qkv_w   = (const float*)d_in[6];
  const float* o_w     = (const float*)d_in[7];
  const float* gate_w  = (const float*)d_in[8];
  const float* up_w    = (const float*)d_in[9];
  const float* down_w  = (const float*)d_in[10];
  const float* sgamma  = (const float*)d_in[11];
  const float* sbeta   = (const float*)d_in[12];
  const float* iter_s  = (const float*)d_in[13];
  const float* q_down  = (const float*)d_in[14];
  const float* q_up    = (const float*)d_in[15];
  const float* k_down  = (const float*)d_in[16];
  const float* k_up    = (const float*)d_in[17];
  const float* route   = (const float*)d_in[18];
  float* out = (float*)d_out;

  // ---- workspace carve ----
  char* p = (char*)d_ws;
  auto alloc = [&](size_t bytes) -> void* {
    void* r = (void*)p;
    p += (bytes + 255) & ~(size_t)255;
    return r;
  };
  __bf16* qkv_wb  = (__bf16*)alloc((size_t)3 * D_ * D_ * 2);
  __bf16* o_wb    = (__bf16*)alloc((size_t)D_ * D_ * 2);
  __bf16* gate_wb = (__bf16*)alloc((size_t)FF_ * D_ * 2);
  __bf16* up_wb   = (__bf16*)alloc((size_t)FF_ * D_ * 2);
  __bf16* down_wb = (__bf16*)alloc((size_t)D_ * FF_ * 2);
  __bf16* lm_wb   = (__bf16*)alloc((size_t)V_ * D_ * 2);
  float*  x       = (float*)alloc((size_t)NTOK * D_ * 4);
  float*  x_in    = (float*)alloc((size_t)NTOK * D_ * 4);
  __bf16* h_bf    = (__bf16*)alloc((size_t)NTOK * D_ * 2);
  float*  qkv_f   = (float*)alloc((size_t)NTOK * 3 * D_ * 4);
  __bf16* qs      = (__bf16*)alloc((size_t)B_ * H_ * T_ * HD_ * 2);
  __bf16* ks      = (__bf16*)alloc((size_t)B_ * H_ * T_ * HD_ * 2);
  __bf16* vt      = (__bf16*)alloc((size_t)B_ * H_ * HD_ * T_ * 2);
  float*  scores  = (float*)alloc((size_t)B_ * H_ * T_ * T_ * 4);
  __bf16* Pm      = (__bf16*)alloc((size_t)B_ * H_ * T_ * T_ * 2);
  float*  attn_o  = (float*)alloc((size_t)B_ * H_ * T_ * HD_ * 4);
  __bf16* o_in    = (__bf16*)alloc((size_t)NTOK * D_ * 2);
  float*  tmp     = (float*)alloc((size_t)NTOK * D_ * 4);
  float*  gate_f  = (float*)alloc((size_t)NTOK * FF_ * 4);
  float*  up_f    = (float*)alloc((size_t)NTOK * FF_ * 4);
  __bf16* ffn_bf  = (__bf16*)alloc((size_t)NTOK * FF_ * 2);
  float*  bqd     = (float*)alloc((size_t)H_ * HD_ * RK * 4);
  float*  bqu     = (float*)alloc((size_t)H_ * RK * HD_ * 4);
  float*  bkd     = (float*)alloc((size_t)H_ * HD_ * RK * 4);
  float*  bku     = (float*)alloc((size_t)H_ * RK * HD_ * 4);

  const int ND = NTOK * D_;          // 1,048,576
  const int NF = NTOK * FF_;         // 4,194,304
  const int BH = B_ * H_;            // 32

  // ---- weight conversion (once per launch; deterministic) ----
  auto cvt = [&](const float* s, __bf16* d, int n) {
    cvt_bf16_kernel<<<ceil_div(n, 256), 256, 0, stream>>>(s, d, n);
  };
  cvt(qkv_w, qkv_wb, 3 * D_ * D_);
  cvt(o_w, o_wb, D_ * D_);
  cvt(gate_w, gate_wb, FF_ * D_);
  cvt(up_w, up_wb, FF_ * D_);
  cvt(down_w, down_wb, D_ * FF_);
  cvt(lm_w, lm_wb, V_ * D_);

  // ---- embedding ----
  embed_kernel<<<ceil_div(ND, 256), 256, 0, stream>>>(tokens, embed_w, x);

  auto gemm = [&](const __bf16* A, int lda, long sA,
                  const __bf16* Bmat, int ldb, long sB,
                  float* C, int ldc, long sC, int M, int N, int K, int batch) {
    if (N % 128 == 0) {
      dim3 grid(N / 128, ceil_div(M, BM), batch);
      gemm_bf16_kernel<128><<<grid, 256, 0, stream>>>(A, lda, sA, Bmat, ldb, sB,
                                                      C, ldc, sC, M, N, K);
    } else {
      dim3 grid(ceil_div(N, 64), ceil_div(M, BM), batch);
      gemm_bf16_kernel<64><<<grid, 256, 0, stream>>>(A, lda, sA, Bmat, ldb, sB,
                                                     C, ldc, sC, M, N, K);
    }
  };

  // ---- layers ----
  for (int l = 0; l < NL; l++) {
    int si = l / ITERS;
    const float* gamma = sgamma + (long)si * D_;
    const float* beta  = sbeta + (long)si * D_;

    // x_in = x
    (void)hipMemcpyAsync(x_in, x, (size_t)ND * 4, hipMemcpyDeviceToDevice, stream);

    // h = bf16(rms(x, norm1)*gamma + beta)
    rmsnorm_kernel<<<NTOK / 8, 256, 0, stream>>>(x, norm1_w, gamma, beta, h_bf);

    // qkv = h @ qkv_w^T  : (1024 x 3072)
    gemm(h_bf, D_, 0, qkv_wb, D_, 0, qkv_f, 3 * D_, 0, NTOK, 3 * D_, D_, 1);

    // blended LoRA modes for this layer
    blendmodes_kernel<<<(4 * H_ * HD_ * RK) / 256, 256, 0, stream>>>(
        route + (long)l * NM, q_down, q_up, k_down, k_up, bqd, bqu, bkd, bku);

    // LoRA delta + bf16 staging of q,k,(v transposed)
    lora_stage_kernel<<<(B_ * H_ * T_) / 8, 256, 0, stream>>>(
        qkv_f, bqd, bqu, bkd, bku, qs, ks, vt);

    // scores = q @ k^T per (b,h) : batch=32, M=N=512, K=64
    gemm(qs, HD_, (long)T_ * HD_, ks, HD_, (long)T_ * HD_,
         scores, T_, (long)T_ * T_, T_, T_, HD_, BH);

    // causal softmax -> P (bf16)
    softmax_kernel<<<(BH * T_) / 8, 256, 0, stream>>>(scores, Pm);

    // attn_out = P @ V : batch=32, M=512, N=64, K=512  (B = v^T staged)
    gemm(Pm, T_, (long)T_ * T_, vt, T_, (long)HD_ * T_,
         attn_o, HD_, (long)T_ * HD_, T_, HD_, T_, BH);

    // merge heads -> bf16 (NTOK, D)
    merge_heads_kernel<<<ceil_div(ND, 256), 256, 0, stream>>>(attn_o, o_in);

    // tmp = o_in @ o_w^T ; x += tmp
    gemm(o_in, D_, 0, o_wb, D_, 0, tmp, D_, 0, NTOK, D_, D_, 1);
    add_kernel<<<ceil_div(ND, 256), 256, 0, stream>>>(x, tmp, ND);

    // h2 = bf16(rms(x, norm2)*gamma + beta)
    rmsnorm_kernel<<<NTOK / 8, 256, 0, stream>>>(x, norm2_w, gamma, beta, h_bf);

    // gate/up : (1024 x 4096)
    gemm(h_bf, D_, 0, gate_wb, D_, 0, gate_f, FF_, 0, NTOK, FF_, D_, 1);
    gemm(h_bf, D_, 0, up_wb, D_, 0, up_f, FF_, 0, NTOK, FF_, D_, 1);
    silu_mul_kernel<<<ceil_div(NF, 256), 256, 0, stream>>>(gate_f, up_f, ffn_bf, NF);

    // tmp = ffn @ down_w^T ; x = x_in + (x + tmp - x_in)*iter
    gemm(ffn_bf, FF_, 0, down_wb, FF_, 0, tmp, D_, 0, NTOK, D_, FF_, 1);
    iterblend_kernel<<<ceil_div(ND, 256), 256, 0, stream>>>(x_in, tmp, x, iter_s, l);
  }

  // ---- final norm + LM head ----
  rmsnorm_kernel<<<NTOK / 8, 256, 0, stream>>>(x, norm_w, nullptr, nullptr, h_bf);
  gemm(h_bf, D_, 0, lm_wb, D_, 0, out, V_, 0, NTOK, V_, D_, 1);
}
